// GPTLanguageModel_18219251270283
// MI455X (gfx1250) — matrix operand order repaired and checked by reading, not compiled, not run on hardware
//
#include <hip/hip_runtime.h>
#include <cstdint>
#include <cstddef>

// ---------- model constants (match reference) ----------
constexpr int V_  = 50257;
constexpr int T_  = 1024;
constexpr int L_  = 6;
constexpr int H_  = 12;
constexpr int D_  = 768;
constexpr int HD_ = 64;
constexpr int B_  = 2;
constexpr int S_  = 1024;
constexpr int M_  = B_ * S_;        // 2048 token rows

// ---------- vector types for WMMA ----------
typedef __attribute__((ext_vector_type(16))) __bf16 v16bf;
typedef __attribute__((ext_vector_type(8)))  __bf16 v8bf;
typedef __attribute__((ext_vector_type(8)))  float  v8f;

// fp32 -> bf16, round-to-nearest-even
static __device__ __forceinline__ __bf16 f2bf(float f) {
    union { float f; uint32_t u; } v; v.f = f;
    uint32_t r = v.u + 0x7FFFu + ((v.u >> 16) & 1u);
    uint16_t h = (uint16_t)(r >> 16);
    return __builtin_bit_cast(__bf16, h);
}

// generic LDS pointer -> 32-bit DS address (low word of the shared aperture)
static __device__ __forceinline__ uint32_t lds_off(const void* p) {
    return (uint32_t)(uintptr_t)p;
}

// CDNA5 async copy: 16 bytes global -> LDS per lane, tracked by ASYNCcnt
static __device__ __forceinline__ void async_cp_b128(uint32_t lds, const void* g) {
    asm volatile("global_load_async_to_lds_b128 %0, %1, off"
                 :: "v"(lds), "v"((uint64_t)(uintptr_t)g)
                 : "memory");
}

// ---------- 1. embedding ----------
__global__ void embed_kernel(const int* __restrict__ idx,
                             const float* __restrict__ tok_emb,
                             const float* __restrict__ pos_emb,
                             float* __restrict__ x) {
    int row = blockIdx.x;
    int s   = row % S_;
    size_t tok = (size_t)idx[row] * D_;
    size_t pos = (size_t)s * D_;
    size_t out = (size_t)row * D_;
    for (int c = threadIdx.x; c < D_; c += blockDim.x)
        x[out + c] = tok_emb[tok + c] + pos_emb[pos + c];
}

// ---------- 2. layernorm (fp32 in, bf16 out) ----------
__global__ void ln_kernel(const float* __restrict__ x,
                          const float* __restrict__ g,
                          const float* __restrict__ b,
                          __bf16* __restrict__ y) {
    __shared__ float rs[256], rq[256];
    int row = blockIdx.x;
    int tid = threadIdx.x;
    size_t base = (size_t)row * D_;
    float s = 0.f, q = 0.f;
    float v0[3];
    #pragma unroll
    for (int i = 0; i < 3; ++i) {
        float v = x[base + tid + i * 256];
        v0[i] = v; s += v; q += v * v;
    }
    rs[tid] = s; rq[tid] = q;
    __syncthreads();
    for (int off = 128; off > 0; off >>= 1) {
        if (tid < off) { rs[tid] += rs[tid + off]; rq[tid] += rq[tid + off]; }
        __syncthreads();
    }
    float mu  = rs[0] * (1.0f / D_);
    float var = rq[0] * (1.0f / D_) - mu * mu;
    float inv = rsqrtf(var + 1e-5f);
    #pragma unroll
    for (int i = 0; i < 3; ++i) {
        int c = tid + i * 256;
        y[base + c] = f2bf((v0[i] - mu) * inv * g[c] + b[c]);
    }
}

// ---------- 3. pack fp32 W[K,N] into per-lane bf16 B-fragment layout ----------
// layout: [nTile][kTile][lane(32)][16 halves]; lane holds col = nTile*16 + (lane&15),
// k = kTile*32 + (lane>>4)*16 + j   (contiguous k order, same as the A fragments: 16 contiguous K per lane)
__global__ void pack_w_kernel(const float* __restrict__ W,
                              __bf16* __restrict__ P,
                              int K, int N, int KT, int total) {
    int e = blockIdx.x * blockDim.x + threadIdx.x;
    if (e >= total) return;
    int j    = e & 15;
    int lane = (e >> 4) & 31;
    int tmp  = e >> 9;
    int kt   = tmp % KT;
    int nt   = tmp / KT;
    int col  = nt * 16 + (lane & 15);
    int k    = kt * 32 + ((lane >> 4) << 4) + j;
    float v  = (col < N) ? W[(size_t)k * N + col] : 0.f;
    P[e] = f2bf(v);
}

// ---------- 4. GEMM with async double-buffered LDS staging ----------
// block = 256 threads = 8 waves; block tile M64 x N256; wave tile M32 x N64.
// Per k-step each lane issues 5 async b128 copies (1 for A, 4 for B).
static __device__ __forceinline__ void stage_tiles(
        const __bf16* __restrict__ A, const __bf16* __restrict__ Wp,
        __bf16* As, __bf16* Bs,
        int m0blk, int ntBase, int kt, int K, int KT, int tid) {
    // A tile: 64 rows x 32 halves (row-major in LDS)
    int r = tid >> 2, c = (tid & 3) << 3;
    async_cp_b128(lds_off(As + r * 32 + c),
                  A + (size_t)(m0blk + r) * K + kt * 32 + c);
    // B tile: 16 n-tiles x 512 halves (packed fragments, contiguous per n-tile)
    #pragma unroll
    for (int i = 0; i < 4; ++i) {
        int idx = tid + (i << 8);          // 0..1023, 16B chunk id
        int ntl = idx >> 6;                // n-tile 0..15
        int rem = (idx & 63) << 3;         // half offset within 512
        async_cp_b128(lds_off(Bs + ntl * 512 + rem),
                      Wp + ((size_t)(ntBase + ntl) * KT + kt) * 512 + rem);
    }
}

__global__ void gemm_bf16_kernel(const __bf16* __restrict__ A,
                                 const __bf16* __restrict__ Wp,
                                 const float* __restrict__ bias,
                                 const float* __restrict__ residual,
                                 float* __restrict__ outF,
                                 __bf16* __restrict__ outB,
                                 int K, int N, int KT, int do_gelu) {
    __shared__ __bf16 As[2][64 * 32];      //  4 KB per stage
    __shared__ __bf16 Bs[2][16 * 512];     // 16 KB per stage   (40 KB total)

    int tid  = threadIdx.x;
    int lane = tid & 31;
    int wid  = tid >> 5;
    int waveM = wid >> 2;                  // 0..1
    int waveN = wid & 3;                   // 0..3
    int m0blk  = blockIdx.y * 64;
    int ntBase = blockIdx.x * 16;          // block covers 16 n-tiles (N=256)

    v8f acc[2][4];
    #pragma unroll
    for (int s = 0; s < 2; ++s)
        #pragma unroll
        for (int t = 0; t < 4; ++t)
            #pragma unroll
            for (int i = 0; i < 8; ++i) acc[s][t][i] = 0.f;

    // prologue: stage k-step 0
    stage_tiles(A, Wp, As[0], Bs[0], m0blk, ntBase, 0, K, KT, tid);

    int koff = (lane >> 4) << 4;           // A fragment k base for this lane (16 contiguous K)
    for (int kt = 0; kt < KT; ++kt) {
        int cur = kt & 1;
        if (kt + 1 < KT) {
            stage_tiles(A, Wp, As[cur ^ 1], Bs[cur ^ 1],
                        m0blk, ntBase, kt + 1, K, KT, tid);
            asm volatile("s_wait_asynccnt 0x5" ::: "memory"); // current stage landed
        } else {
            asm volatile("s_wait_asynccnt 0x0" ::: "memory");
        }
        __syncthreads();

        // A fragments from LDS (two 16B ds loads each)
        v16bf af[2];
        #pragma unroll
        for (int s = 0; s < 2; ++s) {
            const __bf16* ap = As[cur] + (waveM * 32 + s * 16 + (lane & 15)) * 32 + koff;
            v8bf a0 = *(const v8bf*)(ap);
            v8bf a1 = *(const v8bf*)(ap + 8);
            af[s] = __builtin_shufflevector(a0, a1,
                0,1,2,3,4,5,6,7,8,9,10,11,12,13,14,15);
        }
        // B fragments from LDS, 8 WMMAs per k-step per wave
        #pragma unroll
        for (int t = 0; t < 4; ++t) {
            const v16bf bf = *(const v16bf*)(Bs[cur] + (waveN * 4 + t) * 512 + lane * 16);
            #pragma unroll
            for (int s = 0; s < 2; ++s)
                acc[s][t] = __builtin_amdgcn_wmma_f32_16x16x32_bf16(
                    false, af[s], false, bf, (short)0, acc[s][t], false, false);
        }
        __syncthreads();                   // protect buffers before next overwrite
    }

    // epilogue: fuse bias / gelu / residual / store
    int rbase = m0blk + waveM * 32 + ((lane >> 4) << 3);
    #pragma unroll
    for (int s = 0; s < 2; ++s) {
        #pragma unroll
        for (int t = 0; t < 4; ++t) {
            int col = (ntBase + waveN * 4 + t) * 16 + (lane & 15);
            if (col >= N) continue;
            float bv = bias[col];
            #pragma unroll
            for (int i = 0; i < 8; ++i) {
                float v = acc[s][t][i] + bv;
                if (do_gelu) v = 0.5f * v * (1.0f + erff(v * 0.70710678118654752f));
                size_t o = (size_t)(rbase + s * 16 + i) * N + col;
                if (residual) v += residual[o];
                if (outF) outF[o] = v;
                if (outB) outB[o] = f2bf(v);
            }
        }
    }
}

// ---------- 5. fp32 -> bf16 convert ----------
__global__ void cvt_bf16_kernel(const float* __restrict__ in,
                                __bf16* __restrict__ out, int n) {
    int i = blockIdx.x * blockDim.x + threadIdx.x;
    if (i < n) out[i] = f2bf(in[i]);
}

// ---------- 6. causal attention, wave-per-query-row ----------
__global__ void attn_kernel(const float* __restrict__ qkv,
                            float* __restrict__ attn) {
    __shared__ float sc[4][S_];
    __shared__ float qs[4][HD_];
    int lane = threadIdx.x & 31;
    int wid  = threadIdx.x >> 5;
    int bh = blockIdx.x;
    int b = bh / H_, h = bh % H_;
    int qi = blockIdx.y * 4 + wid;

    const float* qrow = qkv + (size_t)(b * S_ + qi) * (3 * D_) + h * HD_;
    for (int d = lane; d < HD_; d += 32) qs[wid][d] = qrow[d];
    __syncthreads();

    float lmax = -INFINITY;
    for (int j = lane; j <= qi; j += 32) {
        const float* krow = qkv + (size_t)(b * S_ + j) * (3 * D_) + D_ + h * HD_;
        float s = 0.f;
        #pragma unroll 8
        for (int d = 0; d < HD_; ++d) s += qs[wid][d] * krow[d];
        s *= 0.125f;
        sc[wid][j] = s;
        lmax = fmaxf(lmax, s);
    }
    #pragma unroll
    for (int off = 16; off > 0; off >>= 1)
        lmax = fmaxf(lmax, __shfl_xor(lmax, off, 32));

    float lsum = 0.f;
    for (int j = lane; j <= qi; j += 32) {
        float p = __expf(sc[wid][j] - lmax);
        sc[wid][j] = p;
        lsum += p;
    }
    #pragma unroll
    for (int off = 16; off > 0; off >>= 1)
        lsum += __shfl_xor(lsum, off, 32);
    __syncthreads();

    int d0 = lane, d1 = lane + 32;
    float o0 = 0.f, o1 = 0.f;
    const float* vbase = qkv + (size_t)(b * S_) * (3 * D_) + 2 * D_ + h * HD_;
    for (int j = 0; j <= qi; ++j) {
        float p = sc[wid][j];
        const float* vrow = vbase + (size_t)j * (3 * D_);
        o0 += p * vrow[d0];
        o1 += p * vrow[d1];
    }
    float inv = 1.0f / lsum;
    size_t ob = (size_t)(b * S_ + qi) * D_ + h * HD_;
    attn[ob + d0] = o0 * inv;
    attn[ob + d1] = o1 * inv;
}

// ---------- host orchestration ----------
static inline int npad256(int n) { return ((n + 255) / 256) * 256; }

extern "C" void kernel_launch(void* const* d_in, const int* in_sizes, int n_in,
                              void* d_out, int out_size, void* d_ws, size_t ws_size,
                              hipStream_t stream) {
    (void)in_sizes; (void)n_in; (void)out_size; (void)ws_size;

    const int*   idx     = (const int*)  d_in[0];
    const float* tok_emb = (const float*)d_in[1];
    const float* pos_emb = (const float*)d_in[2];
    const float* ln1_g   = (const float*)d_in[3];
    const float* ln1_b   = (const float*)d_in[4];
    const float* w_qkv   = (const float*)d_in[5];
    const float* b_qkv   = (const float*)d_in[6];
    const float* w_out   = (const float*)d_in[7];
    const float* b_out   = (const float*)d_in[8];
    const float* ln2_g   = (const float*)d_in[9];
    const float* ln2_b   = (const float*)d_in[10];
    const float* w_fc1   = (const float*)d_in[11];
    const float* b_fc1   = (const float*)d_in[12];
    const float* w_fc2   = (const float*)d_in[13];
    const float* b_fc2   = (const float*)d_in[14];
    const float* lnf_g   = (const float*)d_in[15];
    const float* lnf_b   = (const float*)d_in[16];
    const float* w_head  = (const float*)d_in[17];
    const float* b_head  = (const float*)d_in[18];
    float* logits = (float*)d_out;

    char* ws = (char*)d_ws;
    size_t off = 0;
    float*  x      = (float*) (ws + off); off += (size_t)M_ * D_ * 4;
    __bf16* lnbuf  = (__bf16*)(ws + off); off += (size_t)M_ * D_ * 2;
    float*  qkv    = (float*) (ws + off); off += (size_t)M_ * 3 * D_ * 4;
    float*  attn   = (float*) (ws + off); off += (size_t)M_ * D_ * 4;
    __bf16* attnb  = (__bf16*)(ws + off); off += (size_t)M_ * D_ * 2;
    __bf16* hbuf   = (__bf16*)(ws + off); off += (size_t)M_ * 4 * D_ * 2;
    __bf16* packW  = (__bf16*)(ws + off);

    auto pack = [&](const float* W, int K, int N) {
        int Np = npad256(N);
        int KT = K / 32;
        int total = Np / 16 * KT * 32 * 16;   // == K * Np
        pack_w_kernel<<<(total + 255) / 256, 256, 0, stream>>>(W, packW, K, N, KT, total);
        return Np;
    };
    auto gemm = [&](const __bf16* A, int K, int N, int Np, const float* bias,
                    const float* res, float* oF, __bf16* oB, int gelu) {
        dim3 grid(Np / 256, M_ / 64);
        gemm_bf16_kernel<<<grid, 256, 0, stream>>>(A, packW, bias, res, oF, oB, K, N, K / 32, gelu);
    };

    embed_kernel<<<M_, 256, 0, stream>>>(idx, tok_emb, pos_emb, x);

    for (int l = 0; l < L_; ++l) {
        // ---- attention block ----
        ln_kernel<<<M_, 256, 0, stream>>>(x, ln1_g + (size_t)l * D_, ln1_b + (size_t)l * D_, lnbuf);

        int Np = pack(w_qkv + (size_t)l * D_ * 3 * D_, D_, 3 * D_);
        gemm(lnbuf, D_, 3 * D_, Np, b_qkv + (size_t)l * 3 * D_, nullptr, qkv, nullptr, 0);

        attn_kernel<<<dim3(B_ * H_, S_ / 4), 128, 0, stream>>>(qkv, attn);
        cvt_bf16_kernel<<<(M_ * D_ + 255) / 256, 256, 0, stream>>>(attn, attnb, M_ * D_);

        Np = pack(w_out + (size_t)l * D_ * D_, D_, D_);
        gemm(attnb, D_, D_, Np, b_out + (size_t)l * D_, x, x, nullptr, 0);

        // ---- MLP block ----
        ln_kernel<<<M_, 256, 0, stream>>>(x, ln2_g + (size_t)l * D_, ln2_b + (size_t)l * D_, lnbuf);

        Np = pack(w_fc1 + (size_t)l * D_ * 4 * D_, D_, 4 * D_);
        gemm(lnbuf, D_, 4 * D_, Np, b_fc1 + (size_t)l * 4 * D_, nullptr, nullptr, hbuf, 1);

        Np = pack(w_fc2 + (size_t)l * 4 * D_ * D_, 4 * D_, D_);
        gemm(hbuf, 4 * D_, D_, Np, b_fc2 + (size_t)l * D_, x, x, nullptr, 0);
    }

    ln_kernel<<<M_, 256, 0, stream>>>(x, lnf_g, lnf_b, lnbuf);
    int Np = pack(w_head, D_, V_);
    gemm(lnbuf, D_, V_, Np, b_head, nullptr, logits, nullptr, 0);
}
